// Attention_12292196401203
// MI455X (gfx1250) — compile-verified, matrix operand order checked by reading, not run on hardware
//
#include <hip/hip_runtime.h>

// ---------------------------------------------------------------------------
// Types for CDNA5 WMMA (wave32)
// ---------------------------------------------------------------------------
typedef _Float16 half_t;
typedef half_t v16h __attribute__((ext_vector_type(16)));
typedef half_t v8h  __attribute__((ext_vector_type(8)));
typedef float  v8f  __attribute__((ext_vector_type(8)));
typedef float  v4f  __attribute__((ext_vector_type(4)));

static constexpr int kB = 16, kT = 1024, kS = 2048, kH = 1024;

__device__ __forceinline__ v8f wmma_f16(v16h a, v16h b, v8f c) {
  // D = A(16x32 f16) * B(32x16 f16) + C(16x16 f32)
  return __builtin_amdgcn_wmma_f32_16x16x32_f16(
      /*neg_a=*/false, a, /*neg_b=*/false, b,
      /*c_mod=*/(short)0, c, /*reuse_a=*/false, /*reuse_b=*/false);
}

// Load one 16x32 fragment (A layout; B^T uses the identical per-lane pattern).
// Per ISA 7.12.2: lane<16 holds row lane, K = [kh..kh+7] and [kh+16..kh+23],
// kh = 8*(lane>=16). Two contiguous 16-byte chunks per lane.
template <typename AT>
__device__ __forceinline__ v16h load_frag(const AT* __restrict__ p, int ld) {
  const int lane = threadIdx.x & 31;
  const int m    = lane & 15;
  const int kh   = (lane >> 4) * 8;
  const AT* r = p + (size_t)m * ld + kh;
  if constexpr (__is_same(AT, half_t)) {
    v8h lo = *(const v8h*)(r);
    v8h hi = *(const v8h*)(r + 16);
    return __builtin_shufflevector(lo, hi, 0, 1, 2, 3, 4, 5, 6, 7,
                                   8, 9, 10, 11, 12, 13, 14, 15);
  } else {
    v4f f0 = *(const v4f*)(r);
    v4f f1 = *(const v4f*)(r + 4);
    v4f f2 = *(const v4f*)(r + 16);
    v4f f3 = *(const v4f*)(r + 20);
    v16h v;
#pragma unroll
    for (int i = 0; i < 4; ++i) {
      v[i]      = (half_t)f0[i];
      v[4 + i]  = (half_t)f1[i];
      v[8 + i]  = (half_t)f2[i];
      v[12 + i] = (half_t)f3[i];
    }
    return v;
  }
}

// ---------------------------------------------------------------------------
// Elementwise f32 -> f16 convert
// ---------------------------------------------------------------------------
__global__ void __launch_bounds__(256)
cvt_f32_f16(const float* __restrict__ in, half_t* __restrict__ out, size_t n) {
  size_t i = (size_t)blockIdx.x * blockDim.x + threadIdx.x;
  size_t stride = (size_t)gridDim.x * blockDim.x;
  for (; i < n; i += stride) out[i] = (half_t)in[i];
}

// Transpose+convert: context [B,S,H] f32 -> contextT [B,H,S] f16
__global__ void __launch_bounds__(256)
transpose_cvt(const float* __restrict__ in, half_t* __restrict__ out) {
  __shared__ float t[32][33];
  const int b  = blockIdx.z;
  const int h0 = blockIdx.x * 32;
  const int s0 = blockIdx.y * 32;
  for (int i = threadIdx.y; i < 32; i += 8)
    t[i][threadIdx.x] =
        in[(size_t)b * kS * kH + (size_t)(s0 + i) * kH + h0 + threadIdx.x];
  __syncthreads();
  for (int i = threadIdx.y; i < 32; i += 8)
    out[(size_t)b * kH * kS + (size_t)(h0 + i) * kS + s0 + threadIdx.x] =
        (half_t)t[threadIdx.x][i];
}

// ---------------------------------------------------------------------------
// Generic WMMA GEMM:  C[M,N] = op( A[M,K] * Bt[N,K]^T )
//  - 8 waves (256 thr), wave tile 32x64 (2x4 frags), block tile 64x256
//  - software-pipelined: next B frag / next-K A frags loaded into distinct
//    registers before the WMMA on the current ones (partial loadcnt waits)
//  - AT: float or _Float16 (A converted in registers)
//  - SPLIT: A is concat(A | A2) along K at K1 (for comb = [mix | q])
//  - FINAL: epilogue tanh(x + bias[col]) and f32 output
//  - Bt may be batched: b = rowBase/rowsPerBatch, Bt += b*btBatchStride
// ---------------------------------------------------------------------------
template <typename AT, bool SPLIT, bool FINAL, typename OT>
__global__ void __launch_bounds__(256)
gemm_wmma_k(const AT* __restrict__ A, int lda,
            const AT* __restrict__ A2, int lda2, int K1,
            const half_t* __restrict__ Bt, int ldb,
            long long btBatchStride, int rowsPerBatch,
            const float* __restrict__ bias,
            OT* __restrict__ C, int ldc, int K) {
  const int lane = threadIdx.x & 31;
  const int w    = threadIdx.x >> 5;
  const int wm   = w >> 2;        // 0..1
  const int wn   = w & 3;         // 0..3
  const int rowBase = blockIdx.x * 64 + wm * 32;
  const int colBase = blockIdx.y * 256 + wn * 64;
  const half_t* BtE =
      Bt + (long long)((blockIdx.x * 64) / rowsPerBatch) * btBatchStride;

  const int m  = lane & 15;
  const int kh = (lane >> 4) * 8;

  auto loadArow = [&](int r, int k) -> v16h {
    const AT* Ap; int ldA; int kk;
    if constexpr (SPLIT) {
      if (k < K1) { Ap = A;  ldA = lda;  kk = k; }
      else        { Ap = A2; ldA = lda2; kk = k - K1; }
    } else { Ap = A; ldA = lda; kk = k; }
    return load_frag<AT>(Ap + (size_t)r * ldA + kk, ldA);
  };
  auto loadB = [&](int j, int k) -> v16h {
    return load_frag<half_t>(BtE + (size_t)(colBase + 16 * j) * ldb + k, ldb);
  };

  v8f acc[2][4];
#pragma unroll
  for (int i = 0; i < 2; ++i)
#pragma unroll
    for (int j = 0; j < 4; ++j) acc[i][j] = (v8f){0, 0, 0, 0, 0, 0, 0, 0};

  // pipeline preload (K step 0)
  v16h a0 = loadArow(rowBase, 0);
  v16h a1 = loadArow(rowBase + 16, 0);
  v16h bcur = loadB(0, 0);

  for (int k0 = 0; k0 < K; k0 += 32) {
    const int kn = (k0 + 32 < K) ? k0 + 32 : 0;  // uniform, safe wrap
    v16h a0n, a1n;
#pragma unroll
    for (int j = 0; j < 4; ++j) {
      v16h bnext;
      if (j < 3) {
        bnext = loadB(j + 1, k0);
      } else {
        a0n = loadArow(rowBase, kn);
        a1n = loadArow(rowBase + 16, kn);
        bnext = loadB(0, kn);
      }
      // L2 warm-up two K-tiles ahead
      __builtin_prefetch(
          BtE + (size_t)(colBase + 16 * j + m) * ldb + kh + k0 + 64, 0, 1);
      acc[0][j] = wmma_f16(a0, bcur, acc[0][j]);
      acc[1][j] = wmma_f16(a1, bcur, acc[1][j]);
      bcur = bnext;
    }
    a0 = a0n;
    a1 = a1n;
  }

  // Epilogue. C/D frag layout: lane n=lane&15 is column, VGPR v is row
  // v + 8*(lane>=16) within the 16x16 tile.
  const int n  = lane & 15;
  const int hi = (lane >> 4) * 8;
#pragma unroll
  for (int i = 0; i < 2; ++i)
#pragma unroll
    for (int j = 0; j < 4; ++j) {
      const int c = colBase + 16 * j + n;
      float bv = 0.0f;
      if constexpr (FINAL) bv = bias[c];
#pragma unroll
      for (int v = 0; v < 8; ++v) {
        const int r = rowBase + 16 * i + hi + v;
        float x = acc[i][j][v];
        if constexpr (FINAL)
          C[(size_t)r * ldc + c] = (OT)tanhf(x + bv);
        else
          C[(size_t)r * ldc + c] = (OT)x;
      }
    }
}

// ---------------------------------------------------------------------------
// Fused scores + softmax:
//  block = 512 thr = 16 waves; 16 query rows x all S=2048 keys.
//  wave w owns columns [w*128,+128) -> 8 accum frags (64 VGPRs, leaves room
//  for the double-buffered A/B fragments).
//  softmax in registers + 16x16 LDS cross-wave reduce; probs -> attn (f32).
// ---------------------------------------------------------------------------
__global__ void __launch_bounds__(512)
scores_softmax_k(const half_t* __restrict__ q, const half_t* __restrict__ ctx,
                 float* __restrict__ attn) {
  const int rowBase = blockIdx.x * 16;
  const int b       = rowBase / kT;
  const half_t* ctxb = ctx + (size_t)b * kS * kH;
  const int lane = threadIdx.x & 31;
  const int w    = threadIdx.x >> 5;       // 0..15
  const int colBase = w * 128;
  const int m  = lane & 15;
  const int kh = (lane >> 4) * 8;

  const half_t* qrow = q + (size_t)rowBase * kH;
  auto loadB = [&](int f, int k) -> v16h {
    return load_frag<half_t>(ctxb + (size_t)(colBase + 16 * f) * kH + k, kH);
  };

  v8f acc[8];
#pragma unroll
  for (int f = 0; f < 8; ++f) acc[f] = (v8f){0, 0, 0, 0, 0, 0, 0, 0};

  // pipeline preload
  v16h a_cur = load_frag<half_t>(qrow, kH);
  v16h b_cur = loadB(0, 0);

  for (int k0 = 0; k0 < kH; k0 += 32) {
    const int kn = (k0 + 32 < kH) ? k0 + 32 : 0;
    v16h a_next;
#pragma unroll
    for (int f = 0; f < 8; ++f) {
      v16h b_next;
      if (f < 7) {
        b_next = loadB(f + 1, k0);
      } else {
        a_next = load_frag<half_t>(qrow + kn, kH);
        b_next = loadB(0, kn);
      }
      __builtin_prefetch(
          ctxb + (size_t)(colBase + 16 * f + m) * kH + kh + k0 + 64, 0, 1);
      acc[f] = wmma_f16(a_cur, b_cur, acc[f]);
      b_cur = b_next;
    }
    a_cur = a_next;
  }

  // ---- softmax over the 2048 columns of each of the 16 rows ----
  __shared__ float red[16][16];  // [wave][row]
  const int hi = lane >> 4;      // VGPR v holds row v + 8*hi

  float mx[8];
#pragma unroll
  for (int v = 0; v < 8; ++v) {
    float t = acc[0][v];
#pragma unroll
    for (int f = 1; f < 8; ++f) t = fmaxf(t, acc[f][v]);
    mx[v] = t;
  }
#pragma unroll
  for (int v = 0; v < 8; ++v)
    for (int off = 1; off < 16; off <<= 1)
      mx[v] = fmaxf(mx[v], __shfl_xor(mx[v], off, 32));
  if ((lane & 15) == 0)
#pragma unroll
    for (int v = 0; v < 8; ++v) red[w][hi * 8 + v] = mx[v];
  __syncthreads();
  float gmx[8];
#pragma unroll
  for (int v = 0; v < 8; ++v) {
    float t = -3.4e38f;
    for (int ww = 0; ww < 16; ++ww) t = fmaxf(t, red[ww][hi * 8 + v]);
    gmx[v] = t;
  }
  __syncthreads();

  float sm[8];
#pragma unroll
  for (int v = 0; v < 8; ++v) sm[v] = 0.0f;
#pragma unroll
  for (int f = 0; f < 8; ++f)
#pragma unroll
    for (int v = 0; v < 8; ++v) {
      float e = __expf(acc[f][v] - gmx[v]);
      acc[f][v] = e;
      sm[v] += e;
    }
#pragma unroll
  for (int v = 0; v < 8; ++v)
    for (int off = 1; off < 16; off <<= 1)
      sm[v] += __shfl_xor(sm[v], off, 32);
  if ((lane & 15) == 0)
#pragma unroll
    for (int v = 0; v < 8; ++v) red[w][hi * 8 + v] = sm[v];
  __syncthreads();
  float inv[8];
#pragma unroll
  for (int v = 0; v < 8; ++v) {
    float s = 0.0f;
    for (int ww = 0; ww < 16; ++ww) s += red[ww][hi * 8 + v];
    inv[v] = 1.0f / s;
  }

  // normalized probabilities -> attn (f32 output, consumed later as GEMM-A)
  const int n = lane & 15;
#pragma unroll
  for (int f = 0; f < 8; ++f) {
    const int c = colBase + 16 * f + n;
#pragma unroll
    for (int v = 0; v < 8; ++v) {
      const int r = rowBase + hi * 8 + v;
      attn[(size_t)r * kS + c] = acc[f][v] * inv[v];
    }
  }
}

// ---------------------------------------------------------------------------
// Host launcher
// ---------------------------------------------------------------------------
extern "C" void kernel_launch(void* const* d_in, const int* in_sizes, int n_in,
                              void* d_out, int out_size, void* d_ws,
                              size_t ws_size, hipStream_t stream) {
  (void)in_sizes; (void)n_in; (void)out_size; (void)ws_size;

  const float* output  = (const float*)d_in[0];  // [B,T,H]
  const float* context = (const float*)d_in[1];  // [B,S,H]
  const float* attn_w  = (const float*)d_in[2];  // [H,H]
  const float* lin_w   = (const float*)d_in[3];  // [H,2H]
  const float* lin_b   = (const float*)d_in[4];  // [H]

  float* out_final = (float*)d_out;                         // [B,T,H]
  float* attn_out  = out_final + (size_t)kB * kT * kH;      // [B,T,S]

  auto align256 = [](size_t x) { return (x + 255) & ~(size_t)255; };
  char* p = (char*)d_ws;
  size_t off = 0;
  half_t* aw_h   = (half_t*)(p + off); off += align256((size_t)kH * kH * sizeof(half_t));
  half_t* lw_h   = (half_t*)(p + off); off += align256((size_t)kH * 2 * kH * sizeof(half_t));
  half_t* ctx_h  = (half_t*)(p + off); off += align256((size_t)kB * kS * kH * sizeof(half_t));
  half_t* ctxT_h = (half_t*)(p + off); off += align256((size_t)kB * kH * kS * sizeof(half_t));
  half_t* q_h    = (half_t*)(p + off); off += align256((size_t)kB * kT * kH * sizeof(half_t));
  half_t* mix_h  = (half_t*)(p + off); off += align256((size_t)kB * kT * kH * sizeof(half_t));

  const int M = kB * kT;  // 16384

  // 1) one-time precision conversion / layout prep
  cvt_f32_f16<<<512, 256, 0, stream>>>(attn_w, aw_h, (size_t)kH * kH);
  cvt_f32_f16<<<1024, 256, 0, stream>>>(lin_w, lw_h, (size_t)kH * 2 * kH);
  cvt_f32_f16<<<4096, 256, 0, stream>>>(context, ctx_h, (size_t)kB * kS * kH);
  transpose_cvt<<<dim3(kH / 32, kS / 32, kB), dim3(32, 8), 0, stream>>>(context, ctxT_h);

  // 2) q = output @ attn_w^T   (A f32 in-flight convert, out f16)
  gemm_wmma_k<float, false, false, half_t>
      <<<dim3(M / 64, kH / 256), 256, 0, stream>>>(
          output, kH, nullptr, 0, kH,
          aw_h, kH, 0LL, M, nullptr, q_h, kH, kH);

  // 3) scores = q @ context^T, softmax, write attn (f32 output #2)
  scores_softmax_k<<<M / 16, 512, 0, stream>>>(q_h, ctx_h, attn_out);

  // 4) mix = attn @ context    (A = f32 attn from d_out, batched B = ctxT)
  gemm_wmma_k<float, false, false, half_t>
      <<<dim3(M / 64, kH / 256), 256, 0, stream>>>(
          attn_out, kS, nullptr, 0, kS,
          ctxT_h, kS, (long long)kH * kS, kT, nullptr, mix_h, kH, kS);

  // 5) out = tanh([mix|q] @ lin_w^T + b)  (split-K A, fused epilogue, f32 out)
  gemm_wmma_k<half_t, true, true, float>
      <<<dim3(M / 64, kH / 256), 256, 0, stream>>>(
          mix_h, kH, q_h, kH, kH,
          lw_h, 2 * kH, 0LL, M, lin_b, out_final, kH, 2 * kH);
}
